// AdaSoftmaxGenerator_26903675142504
// MI455X (gfx1250) — matrix operand order repaired and checked by reading, not compiled, not run on hardware
//
#include <hip/hip_runtime.h>
#include <hip/hip_bf16.h>

typedef __bf16 v16bf __attribute__((ext_vector_type(16)));
typedef __bf16 v8bf  __attribute__((ext_vector_type(8)));
typedef float  v8f   __attribute__((ext_vector_type(8)));
typedef unsigned int u32x4 __attribute__((ext_vector_type(4)));
typedef int          i32x4 __attribute__((ext_vector_type(4)));
typedef int          i32x8 __attribute__((ext_vector_type(8)));

#define D_DIM      1024
#define B_DIM      1024
#define OUT_STRIDE 128000
#define NEG_BIG   -3.0e38f
#define LDS_ROW    36            // 32 data dwords + 4 pad dwords (TDM pad_enable)

// ---------------------------------------------------------------------------
// 1) Convert x fp32 -> bf16 (2 MB, stays hot in the 192 MB L2)
// ---------------------------------------------------------------------------
__global__ __launch_bounds__(256) void cvt_x_bf16(const float* __restrict__ x,
                                                  __bf16* __restrict__ xb, int n) {
    for (int i = blockIdx.x * 256 + threadIdx.x; i < n; i += gridDim.x * 256)
        xb[i] = (__bf16)x[i];
}

// ---------------------------------------------------------------------------
// TDM: DMA a 128x32 fp32 W tile (rows o0..o0+127, cols k0..k0+31) into LDS.
// pad_enable inserts 4 pad dwords after every 32 data dwords -> LDS row stride
// 36 dwords. Rows beyond the tensor (o >= O) are zero-filled by hardware.
// D# layout per CDNA5 ISA ch.8 (group0 128b, group1 256b; groups 2/3 unused).
// ---------------------------------------------------------------------------
__device__ inline void tdm_load_tile(const float* W, unsigned lds_off,
                                     int o0, int O, int k0) {
    unsigned long long ga =
        (unsigned long long)(uintptr_t)(W + (size_t)o0 * D_DIM + k0);
    unsigned rows_left = (unsigned)(O - o0);          // tensor_dim1 from tile start

    u32x4 g0;
    g0[0] = 1u;                                       // count=1, user descriptor
    g0[1] = lds_off;                                  // lds_addr (bytes)
    g0[2] = (unsigned)(ga & 0xFFFFFFFFu);             // global_addr[31:0]
    g0[3] = (unsigned)((ga >> 32) & 0x01FFFFFFu)      // global_addr[56:32]
          | (2u << 30);                               // type=2 ("image")

    i32x8 g1;
    g1[0] = (2 << 16)                                 // data_size = 4 bytes
          | (1 << 20)                                 // pad_enable
          | (4 << 22)                                 // pad_interval: every 32 dwords
          | (3 << 25);                                // pad_amount: 4 dwords
    g1[1] = (int)((1024u & 0xFFFFu) << 16);           // tensor_dim0 lo (K extent)
    g1[2] = (int)((rows_left & 0xFFFFu) << 16);       // dim0 hi(=0) | tensor_dim1 lo
    g1[3] = (int)(((rows_left >> 16) & 0xFFFFu)       // tensor_dim1 hi
          | (32u << 16));                             // tile_dim0 = 32 elements
    g1[4] = 128;                                      // tile_dim1 = 128 rows
    g1[5] = 1024;                                     // tensor_dim0_stride
    g1[6] = 0;
    g1[7] = 0;

    i32x4 z4 = {0, 0, 0, 0};
#if defined(__clang_major__) && (__clang_major__ >= 23)
    i32x8 z8 = {0, 0, 0, 0, 0, 0, 0, 0};
    __builtin_amdgcn_tensor_load_to_lds(g0, g1, z4, z4, z8, 0);
#else
    __builtin_amdgcn_tensor_load_to_lds(g0, g1, z4, z4, 0);
#endif
}

// ---------------------------------------------------------------------------
// 2) GEMM: out[m, col_off + o] = sum_k xb[m,k] * W[o,k] + bias[o]
//    Block tile 128x128x32, 8 waves, wave = 64x32 = 4x2 WMMA f32_16x16x32_bf16.
//    W tiles arrive via double-buffered TENSOR_LOAD_TO_LDS (fp32, padded rows);
//    fp32->bf16 conversion happens at B-fragment load time.
// ---------------------------------------------------------------------------
__global__ __launch_bounds__(256) void gemm_bf16_wmma(
    const __bf16* __restrict__ xb, const float* __restrict__ W,
    const float* __restrict__ bias, float* __restrict__ out,
    int O, int col_off, int split, float* __restrict__ extra) {

    __shared__ float sW[2][128 * LDS_ROW];            // 2 x 18 KB fp32 tiles

    const int tid  = threadIdx.x;
    const int wave = tid >> 5;
    const int lane = tid & 31;
    const int lm   = lane & 15;
    const int halfSel = lane >> 4;                    // 0: lanes 0-15, 1: 16-31

    const int m0 = blockIdx.y * 128;
    const int o0 = blockIdx.x * 128;
    const int wm = (wave >> 2) * 64;                  // wave M offset: 0 / 64
    const int wn = (wave & 3) * 32;                   // wave N offset: 0..96

    v8f acc[4][2];
    #pragma unroll
    for (int mt = 0; mt < 4; ++mt)
        #pragma unroll
        for (int nt = 0; nt < 2; ++nt)
            #pragma unroll
            for (int r = 0; r < 8; ++r) acc[mt][nt][r] = 0.0f;

    const unsigned lds0 = (unsigned)(uintptr_t)(void*)&sW[0][0];
    const unsigned lds1 = (unsigned)(uintptr_t)(void*)&sW[1][0];

    if (wave == 0) tdm_load_tile(W, lds0, o0, O, 0);

    for (int i = 0; i < 32; ++i) {
        const int cur = i & 1;
        __syncthreads();                              // prev reads of buf[1-cur] done
        if (wave == 0) {
            if (i + 1 < 32) {
                tdm_load_tile(W, cur ? lds0 : lds1, o0, O, (i + 1) * 32);
                __builtin_amdgcn_s_wait_tensorcnt(1); // tile i landed in LDS
            } else {
                __builtin_amdgcn_s_wait_tensorcnt(0);
            }
        }
        __syncthreads();                              // buf[cur] visible to all

        const int k0 = i * 32;
        const float* buf = sW[cur];

        // A fragments straight from global bf16 x (16-bit A 16x32 layout:
        // lanes0-15 hold K {0..7,16..23}, lanes16-31 hold K {8..15,24..31})
        v16bf afrag[4];
        #pragma unroll
        for (int mt = 0; mt < 4; ++mt) {
            const int m = m0 + wm + mt * 16 + lm;
            const __bf16* xp = xb + (size_t)m * D_DIM + k0 + halfSel * 8;
            v8bf lo = *(const v8bf*)(xp);
            v8bf hi = *(const v8bf*)(xp + 16);
            #pragma unroll
            for (int j = 0; j < 8; ++j) { afrag[mt][j] = lo[j]; afrag[mt][8 + j] = hi[j]; }
        }

        // B fragments from LDS fp32 tile (lane = N; lanes0-15 K {0..7,16..23}, lanes16-31 K {8..15,24..31})
        v16bf bfrag[2];
        #pragma unroll
        for (int nt = 0; nt < 2; ++nt) {
            const float* bp = buf + (wn + nt * 16 + lm) * LDS_ROW + halfSel * 8;
            #pragma unroll
            for (int j = 0; j < 16; ++j) bfrag[nt][j] = (__bf16)bp[j < 8 ? j : j + 8];
        }

        #pragma unroll
        for (int mt = 0; mt < 4; ++mt)
            #pragma unroll
            for (int nt = 0; nt < 2; ++nt)
                acc[mt][nt] = __builtin_amdgcn_wmma_f32_16x16x32_bf16(
                    false, afrag[mt], false, bfrag[nt],
                    (short)0, acc[mt][nt], false, false);
    }

    // epilogue: D layout -> VGPR r: lanes0-15 M=r, lanes16-31 M=8+r, N=lane%16
    #pragma unroll
    for (int nt = 0; nt < 2; ++nt) {
        const int o = o0 + wn + nt * 16 + lm;
        if (o >= O) continue;
        const float b = bias[o];
        #pragma unroll
        for (int mt = 0; mt < 4; ++mt) {
            #pragma unroll
            for (int r = 0; r < 8; ++r) {
                const int m = m0 + wm + mt * 16 + (halfSel ? r + 8 : r);
                const float v = acc[mt][nt][r] + b;
                if (o < split) out[(size_t)m * OUT_STRIDE + col_off + o] = v;
                else           extra[m * 2 + (o - split)] = v;
            }
        }
    }
}

// ---------------------------------------------------------------------------
// Online (m, s) reduction helpers: s = sum(exp(v - m)) with running max.
// ---------------------------------------------------------------------------
__device__ inline void onlineUpd(float& m, float& s, float v) {
    const float nm = fmaxf(m, v);
    s = s * __expf(m - nm) + __expf(v - nm);
    m = nm;
}
__device__ inline void onlineMerge(float& m, float& s, float m2, float s2) {
    const float nm = fmaxf(m, m2);
    s = s * __expf(m - nm) + s2 * __expf(m2 - nm);
    m = nm;
}
__device__ inline void blockMS(float& m, float& s, float* bm, float* bs) {
    #pragma unroll
    for (int o = 16; o > 0; o >>= 1)
        onlineMerge(m, s, __shfl_xor(m, o, 32), __shfl_xor(s, o, 32));
    const int w = threadIdx.x >> 5, l = threadIdx.x & 31;
    if (l == 0) { bm[w] = m; bs[w] = s; }
    __syncthreads();
    if (w == 0) {
        float tm = (l < 8) ? bm[l] : NEG_BIG;
        float ts = (l < 8) ? bs[l] : 0.0f;
        #pragma unroll
        for (int o = 16; o > 0; o >>= 1)
            onlineMerge(tm, ts, __shfl_xor(tm, o, 32), __shfl_xor(ts, o, 32));
        if (l == 0) { bm[0] = tm; bs[0] = ts; }
    }
    __syncthreads();
    m = bm[0]; s = bs[0];
    __syncthreads();
}

// ---------------------------------------------------------------------------
// 3) Head log-softmax over 20002 logits/row (20000 in out, 2 in extra).
//    Single online read pass + one write pass. Rewrites extra with log-probs.
// ---------------------------------------------------------------------------
__global__ __launch_bounds__(256) void head_lsm(float* __restrict__ out,
                                                float* __restrict__ extra) {
    __shared__ float bm[8], bs[8];
    const int r = blockIdx.x;
    float4* row = (float4*)(out + (size_t)r * OUT_STRIDE);
    float* ex   = extra + r * 2;

    float m = NEG_BIG, s = 0.0f;
    for (int c = threadIdx.x; c < 5000; c += 256) {
        const float4 v = row[c];
        onlineUpd(m, s, v.x); onlineUpd(m, s, v.y);
        onlineUpd(m, s, v.z); onlineUpd(m, s, v.w);
    }
    if (threadIdx.x < 2) onlineUpd(m, s, ex[threadIdx.x]);
    blockMS(m, s, bm, bs);
    const float L = m + __logf(s);

    for (int c = threadIdx.x; c < 5000; c += 256) {
        float4 v = row[c];
        v.x -= L; v.y -= L; v.z -= L; v.w -= L;
        row[c] = v;
    }
    if (threadIdx.x < 2) ex[threadIdx.x] -= L;
}

// ---------------------------------------------------------------------------
// 4) Tail log-softmax (blockIdx.y = segment), adds head log-prob buffer
// ---------------------------------------------------------------------------
__global__ __launch_bounds__(256) void tail_lsm(float* __restrict__ out,
                                                const float* __restrict__ extra) {
    __shared__ float bm[8], bs[8];
    const int r   = blockIdx.x;
    const int seg = blockIdx.y;
    const int off = seg ? 60000 : 20000;
    const int nv4 = seg ? 17000 : 10000;              // O/4
    float4* row = (float4*)(out + (size_t)r * OUT_STRIDE + off);
    const float buffer = extra[r * 2 + seg];

    float m = NEG_BIG, s = 0.0f;
    for (int c = threadIdx.x; c < nv4; c += 256) {
        const float4 v = row[c];
        onlineUpd(m, s, v.x); onlineUpd(m, s, v.y);
        onlineUpd(m, s, v.z); onlineUpd(m, s, v.w);
    }
    blockMS(m, s, bm, bs);
    const float L = m + __logf(s) - buffer;

    for (int c = threadIdx.x; c < nv4; c += 256) {
        float4 v = row[c];
        v.x -= L; v.y -= L; v.z -= L; v.w -= L;
        row[c] = v;
    }
}

// ---------------------------------------------------------------------------
extern "C" void kernel_launch(void* const* d_in, const int* in_sizes, int n_in,
                              void* d_out, int out_size, void* d_ws, size_t ws_size,
                              hipStream_t stream) {
    const float* x      = (const float*)d_in[0];
    const float* head_w = (const float*)d_in[1];
    const float* head_b = (const float*)d_in[2];
    const float* t0_w   = (const float*)d_in[3];
    const float* t0_b   = (const float*)d_in[4];
    const float* t1_w   = (const float*)d_in[5];
    const float* t1_b   = (const float*)d_in[6];
    float* out = (float*)d_out;

    __bf16* xb   = (__bf16*)d_ws;                                     // 2 MB
    float* extra = (float*)((char*)d_ws + (size_t)B_DIM * D_DIM * 2); // 8 KB

    cvt_x_bf16<<<1024, 256, 0, stream>>>(x, xb, B_DIM * D_DIM);

    const dim3 blk(256);
    const int mblocks = B_DIM / 128;
    // head: O=20002, split at 20000 (last 2 columns -> extra buffer)
    gemm_bf16_wmma<<<dim3((20002 + 127) / 128, mblocks), blk, 0, stream>>>(
        xb, head_w, head_b, out, 20002, 0, 20000, extra);
    // tail0: O=40000 -> out columns [20000, 60000)
    gemm_bf16_wmma<<<dim3((40000 + 127) / 128, mblocks), blk, 0, stream>>>(
        xb, t0_w, t0_b, out, 40000, 20000, 40000, extra);
    // tail1: O=68000 -> out columns [60000, 128000)
    gemm_bf16_wmma<<<dim3((68000 + 127) / 128, mblocks), blk, 0, stream>>>(
        xb, t1_w, t1_b, out, 68000, 60000, 68000, extra);

    head_lsm<<<B_DIM, 256, 0, stream>>>(out, extra);
    tail_lsm<<<dim3(B_DIM, 2), 256, 0, stream>>>(out, extra);
}
